// CTLSTM_62818191671998
// MI455X (gfx1250) — compile-verified, matrix operand order checked by reading, not run on hardware
//
#include <hip/hip_runtime.h>
#include <hip/hip_bf16.h>
#include <math.h>

// ---------------- problem constants ----------------
#define HID   1024
#define SEQT  4096
#define G7    (7 * HID)          // 7168
#define NEMB  33                 // TYPES+1 rows in embedding table

typedef __bf16 bf16_t;
typedef __attribute__((ext_vector_type(16))) __bf16 v16bf;
typedef __attribute__((ext_vector_type(8)))  float  v8f;

// ---------------- workspace layout (bytes) ----------------
static constexpr size_t PRE_BYTES  = (size_t)SEQT * G7 * 4;     // 117,440,512
static constexpr size_t WTOP_BYTES = (size_t)G7 * HID * 2;      //  14,680,064
static constexpr size_t EMBB_BYTES = (size_t)NEMB * HID * 2;    //      67,584
static constexpr size_t HBUF_BYTES = (size_t)2 * HID * 4;       //       8,192
static constexpr size_t SYNC_BYTES = 256;
static constexpr size_t OFF_PRE  = 0;
static constexpr size_t OFF_WTOP = OFF_PRE  + PRE_BYTES;
static constexpr size_t OFF_EMBB = OFF_WTOP + WTOP_BYTES;
static constexpr size_t OFF_HBUF = OFF_EMBB + EMBB_BYTES;
static constexpr size_t OFF_SYNC = OFF_HBUF + HBUF_BYTES;
static constexpr size_t WS_NEEDED = OFF_SYNC + SYNC_BYTES;

// =====================================================================
// Kernel 1: transpose+convert W_top (fp32 rows 0..H-1 of [2H][7H]) into
// bf16 [7H][H] so K is contiguous per gate column; convert embedding.
// =====================================================================
__global__ __launch_bounds__(256) void ctlstm_convert(
    const float* __restrict__ w_r, const float* __restrict__ emb,
    bf16_t* __restrict__ wtop, bf16_t* __restrict__ embb) {
  size_t i = (size_t)blockIdx.x * blockDim.x + threadIdx.x;
  const size_t NW = (size_t)G7 * HID;
  if (i < NW) {
    int n = (int)(i >> 10);
    int k = (int)(i & 1023);
    wtop[i] = (bf16_t)w_r[(size_t)k * G7 + n];
  } else {
    size_t j = i - NW;
    if (j < (size_t)NEMB * HID) embb[j] = (bf16_t)emb[j];
  }
}

// =====================================================================
// Kernel 2: pre_gates[T][7H] = gather(emb_bf16, event) @ W_top + b
// Register-blocked: each wave owns a 32x32 output block = 4 WMMA tiles
// (2 A-tiles x 2 B-tiles) -> 8 b128 loads per 4 v_wmma_f32_16x16x32_bf16
// per 32-K step (2x the arithmetic intensity of a 16x16-per-wave scheme).
// 16-bit A/B lane packing per ISA 7.12.2: lane<16 holds row M=lane with
// halves K{kb..kb+7, kb+16..kb+23}; lane>=16 same row, K shifted by 8.
// =====================================================================
__device__ __forceinline__ v16bf pack16(const bf16_t* __restrict__ row,
                                        int k0, int hi) {
  union Chunk { uint4 u; bf16_t h[8]; } lo, up;
  lo.u = *(const uint4*)(row + k0 + hi * 8);
  up.u = *(const uint4*)(row + k0 + 16 + hi * 8);
  v16bf v;
#pragma unroll
  for (int i = 0; i < 8; ++i) { v[i] = lo.h[i]; v[i + 8] = up.h[i]; }
  return v;
}

__global__ __launch_bounds__(256) void ctlstm_gemm_pre(
    const bf16_t* __restrict__ wtop, const bf16_t* __restrict__ embb,
    const int* __restrict__ event, const float* __restrict__ b_r,
    float* __restrict__ pre) {
  const int lane = threadIdx.x & 31;
  const int wv   = threadIdx.x >> 5;
  const int tn   = blockIdx.x * 32;                 // grid.x = 224
  const int tm   = (blockIdx.y * 8 + wv) * 32;      // grid.y = 16, 8 waves/blk
  const int r    = lane & 15;
  const int hi   = lane >> 4;

  const bf16_t* __restrict__ Arow0 = embb + (size_t)event[tm + r] * HID;
  const bf16_t* __restrict__ Arow1 = embb + (size_t)event[tm + 16 + r] * HID;
  const bf16_t* __restrict__ Brow0 = wtop + (size_t)(tn + r) * HID;
  const bf16_t* __restrict__ Brow1 = wtop + (size_t)(tn + 16 + r) * HID;

  v8f acc00 = {}, acc01 = {}, acc10 = {}, acc11 = {};
  for (int k0 = 0; k0 < HID; k0 += 32) {
    __builtin_prefetch((const void*)(Brow0 + k0 + 128), 0, 1);
    __builtin_prefetch((const void*)(Brow1 + k0 + 128), 0, 1);
    v16bf a0 = pack16(Arow0, k0, hi);
    v16bf a1 = pack16(Arow1, k0, hi);
    v16bf b0 = pack16(Brow0, k0, hi);
    v16bf b1 = pack16(Brow1, k0, hi);
    acc00 = __builtin_amdgcn_wmma_f32_16x16x32_bf16(
        false, a0, false, b0, (short)0, acc00, false, false);
    acc01 = __builtin_amdgcn_wmma_f32_16x16x32_bf16(
        false, a0, false, b1, (short)0, acc01, false, false);
    acc10 = __builtin_amdgcn_wmma_f32_16x16x32_bf16(
        false, a1, false, b0, (short)0, acc10, false, false);
    acc11 = __builtin_amdgcn_wmma_f32_16x16x32_bf16(
        false, a1, false, b1, (short)0, acc11, false, false);
  }

  // C/D layout: VGPR rr -> M = rr + 8*hi ; N = lane&15
  const int n0 = tn + (lane & 15);
  const int n1 = n0 + 16;
  const float bias0 = b_r[n0];
  const float bias1 = b_r[n1];
#pragma unroll
  for (int rr = 0; rr < 8; ++rr) {
    const int m0 = tm + rr + 8 * hi;
    const int m1 = m0 + 16;
    pre[(size_t)m0 * G7 + n0] = acc00[rr] + bias0;
    pre[(size_t)m0 * G7 + n1] = acc01[rr] + bias1;
    pre[(size_t)m1 * G7 + n0] = acc10[rr] + bias0;
    pre[(size_t)m1 * G7 + n1] = acc11[rr] + bias1;
  }
}

// =====================================================================
// Kernel 3: persistent sequential scan. 128 WGs x 512 threads.
// WG b owns hidden units [8b, 8b+8): all 7 gate columns for those units
// resident in LDS as fp32 (56 cols x 1024, stride 1025 vs bank clash).
// One grid barrier per step; h_d ping-pongs through global/L2 (4KB).
// =====================================================================
#define NBLK 128
#define TPB  512
#define WSTRIDE 1025

__device__ __forceinline__ float sigm_f(float x) {
  return 1.0f / (1.0f + __expf(-x));
}
__device__ __forceinline__ float softplus_f(float x) {
  return x > 20.0f ? x : log1pf(__expf(x));
}

__global__ __launch_bounds__(TPB) void ctlstm_scan(
    const float* __restrict__ w_r,      // fp32 [2H][7H]; recurrent rows H..2H-1
    const float* __restrict__ pre,      // [T][7H]
    const float* __restrict__ dur,      // [T]
    float* __restrict__ out,            // [5][T][H]
    float* __restrict__ hbuf,           // [2][H]
    unsigned* __restrict__ sync) {
  extern __shared__ float smem[];
  float* w_lds = smem;                       // 56 * 1025
  float* h_lds = w_lds + 56 * WSTRIDE;       // 1024
  float* part  = h_lds + HID;                // 56 * 8
  float* gat   = part + 56 * 8;              // 56

  const int tid  = threadIdx.x;
  const int base = blockIdx.x * 8;

  // one-time: stage this WG's 56 recurrent columns into LDS (fp32)
  for (int idx = tid; idx < 56 * 1024; idx += TPB) {
    int lc = idx >> 10, k = idx & 1023;
    int g = lc >> 3, jj = lc & 7;
    w_lds[lc * WSTRIDE + k] = w_r[(size_t)(HID + k) * G7 + g * HID + base + jj];
  }

  float carry_c = 0.0f, carry_cb = 0.0f;     // live in threads 0..7 only
  unsigned lg = 0;
  unsigned* cnt = sync;
  unsigned* gen = sync + 1;
  __syncthreads();

  for (int t = 0; t < SEQT; ++t) {
    const float* hp = hbuf + (t & 1) * HID;
    for (int i = tid; i < HID; i += TPB) h_lds[i] = hp[i];
    __syncthreads();

    // 56 dot products, K split 8 ways (448 workers)
    const int c = tid >> 3, s = tid & 7;
    if (c < 56) {
      const float* w  = w_lds + c * WSTRIDE + s * 128;
      const float* hh = h_lds + s * 128;
      float sum = 0.0f;
#pragma unroll 8
      for (int i = 0; i < 128; ++i) sum = fmaf(w[i], hh[i], sum);
      part[c * 8 + s] = sum;
    }
    __syncthreads();

    if (tid < 56) {
      int g = tid >> 3, jj = tid & 7;
      float v = pre[(size_t)t * G7 + g * HID + base + jj];
#pragma unroll
      for (int s2 = 0; s2 < 8; ++s2) v += part[tid * 8 + s2];
      gat[tid] = v;
    }
    __syncthreads();

    if (tid < 8) {
      const float gi  = sigm_f(gat[0 * 8 + tid]);
      const float gf  = sigm_f(gat[1 * 8 + tid]);
      const float gz  = tanhf(gat[2 * 8 + tid]);
      const float go  = sigm_f(gat[3 * 8 + tid]);
      const float gib = sigm_f(gat[4 * 8 + tid]);
      const float gfb = sigm_f(gat[5 * 8 + tid]);
      const float gd  = softplus_f(gat[6 * 8 + tid]);

      const float ct  = gf * carry_c + gi * gz;
      const float cbt = gfb * carry_cb + gib * gz;
      const float ht  = go * tanhf(ct);
      const float cd  = cbt + (ct - cbt) * __expf(-gd * dur[t]);
      const float hd  = go * tanhf(cd);
      carry_c = ct; carry_cb = cbt;

      const size_t o = (size_t)t * HID + base + tid;
      const size_t plane = (size_t)SEQT * HID;
      out[o]             = ht;
      out[plane + o]     = ct;
      out[2 * plane + o] = cbt;
      out[3 * plane + o] = go;
      out[4 * plane + o] = gd;
      hbuf[((t + 1) & 1) * HID + base + tid] = hd;
    }

    // grid-wide barrier (generation counter in L2)
    __threadfence();
    __syncthreads();
    if (tid == 0) {
      if (atomicAdd(cnt, 1u) == NBLK - 1) {
        atomicExch(cnt, 0u);
        __threadfence();
        atomicAdd(gen, 1u);
      } else {
        while (__hip_atomic_load(gen, __ATOMIC_RELAXED,
                                 __HIP_MEMORY_SCOPE_AGENT) <= lg) {
          __builtin_amdgcn_s_sleep(1);
        }
      }
    }
    lg++;
    __syncthreads();
  }
}

// =====================================================================
extern "C" void kernel_launch(void* const* d_in, const int* in_sizes, int n_in,
                              void* d_out, int out_size, void* d_ws, size_t ws_size,
                              hipStream_t stream) {
  (void)in_sizes; (void)n_in; (void)out_size;
  if (ws_size < WS_NEEDED) return;

  const int*   event = (const int*)d_in[0];
  const float* dur   = (const float*)d_in[1];
  const float* w_r   = (const float*)d_in[2];
  const float* b_r   = (const float*)d_in[3];
  const float* emb   = (const float*)d_in[4];
  float* out = (float*)d_out;

  char* ws = (char*)d_ws;
  float*    pre  = (float*)(ws + OFF_PRE);
  bf16_t*   wtop = (bf16_t*)(ws + OFF_WTOP);
  bf16_t*   embb = (bf16_t*)(ws + OFF_EMBB);
  float*    hbuf = (float*)(ws + OFF_HBUF);
  unsigned* sync = (unsigned*)(ws + OFF_SYNC);

  hipMemsetAsync(ws + OFF_HBUF, 0, HBUF_BYTES + SYNC_BYTES, stream);

  {
    size_t total = (size_t)G7 * HID + (size_t)NEMB * HID;
    int blocks = (int)((total + 255) / 256);
    ctlstm_convert<<<blocks, 256, 0, stream>>>(w_r, emb, wtop, embb);
  }
  {
    dim3 grid(G7 / 32, SEQT / (32 * 8));   // 224 x 16, 8 waves/block
    ctlstm_gemm_pre<<<grid, 256, 0, stream>>>(wtop, embb, event, b_r, pre);
  }
  {
    size_t smem_bytes = (size_t)(56 * WSTRIDE + HID + 56 * 8 + 56) * sizeof(float);
    hipFuncSetAttribute((const void*)ctlstm_scan,
                        hipFuncAttributeMaxDynamicSharedMemorySize,
                        (int)smem_bytes);
    ctlstm_scan<<<NBLK, TPB, smem_bytes, stream>>>(w_r, pre, dur, out, hbuf, sync);
  }
}
